// DirNet_24223615550107
// MI455X (gfx1250) — matrix operand order repaired and checked by reading, not compiled, not run on hardware
//
#include <hip/hip_runtime.h>
#include <hip/hip_bf16.h>
#include <stdint.h>

typedef __attribute__((ext_vector_type(16))) __bf16 v16bf;
typedef __attribute__((ext_vector_type(8)))  float  v8f;
typedef uint32_t u32x4 __attribute__((ext_vector_type(4), aligned(16)));
typedef uint32_t u32x8 __attribute__((ext_vector_type(8), aligned(16)));

#define BM 128
#define BN 128
#define BK 64
#define PITCH 36   // uint32 per LDS row = 72 bf16 (64 data + 8 pad) = 144 B

// Single-instruction fp32x2 -> packed bf16 (RNE), avoids subregister shuffles.
__device__ __forceinline__ uint32_t pack_bf16(float a, float b) {
    uint32_t r;
    asm("v_cvt_pk_bf16_f32 %0, %1, %2" : "=v"(r) : "v"(a), "v"(b));
    return r;
}

__global__ __launch_bounds__(256)
void dirnet_bf16_wmma(const float* __restrict__ x, const float* __restrict__ W,
                      const float* __restrict__ b, const int* __restrict__ idx,
                      float* __restrict__ out)
{
    constexpr float SCALE = 0.04419417382415922f;   // 1/sqrt(512) * LR_MUL
    constexpr int OSTR = 18 * 512;                  // o stride in x
    constexpr int KSTR = 18 * 18 * 512;             // batch (k) stride in x

    const int tid = threadIdx.x;
    const int bid = blockIdx.x;
    const int o   = bid / (36 * 4);
    const int rem = bid % (36 * 4);
    const int m0  = (rem >> 2) * BM;                // 36 M-tiles (4608/128)
    const int n0  = (rem & 3)  * BN;                // 4 N-tiles (512/128)

    __shared__ int sIdx[18];
    __shared__ __align__(16) uint32_t As[2][BM * PITCH];
    __shared__ __align__(16) uint32_t Bs[2][BN * PITCH];

    if (tid < 18) sIdx[tid] = idx[tid];
    __syncthreads();

    const float* Wo = W + (size_t)o * (512 * 512);

    const int lane = tid & 31;
    const int lo   = lane & 15;
    const int hi   = lane >> 4;
    const int wv   = tid >> 5;     // wave 0..7
    const int wm   = wv >> 2;      // 0..1 : 64-row slab
    const int wn   = wv & 3;       // 0..3 : 32-col slab

    // ---- per-thread staging pointers (fixed; K offset is an immediate) ----
    const float* xp[8];
    const float* wp[8];
    int sOfs[8];
    #pragma unroll
    for (int i = 0; i < 8; ++i) {
        int f   = tid + i * 256;          // float4 slot 0..2047
        int row = f >> 4;                 // 0..127
        int c4  = f & 15;                 // float4 column within 64-float stage row
        int rg  = m0 + row;
        int kk  = rg / 18;
        int jj  = rg - kk * 18;
        int jg  = sIdx[jj];
        xp[i]   = x + (size_t)kk * KSTR + (size_t)o * OSTR + (size_t)jg * 512 + c4 * 4;
        wp[i]   = Wo + (size_t)(n0 + row) * 512 + c4 * 4;
        sOfs[i] = row * PITCH + c4 * 2;
    }

    float4 xr[8], wr[8];

    auto loadStage = [&](int k0) {      // k0 is a compile-time constant after unroll
        #pragma unroll
        for (int i = 0; i < 8; ++i) {
            xr[i] = *(const float4*)(xp[i] + k0);
            wr[i] = *(const float4*)(wp[i] + k0);
        }
    };
    auto storeStage = [&](int buf) {
        #pragma unroll
        for (int i = 0; i < 8; ++i) {
            uint32_t* dA = &As[buf][sOfs[i]];
            dA[0] = pack_bf16(xr[i].x, xr[i].y);
            dA[1] = pack_bf16(xr[i].z, xr[i].w);
            uint32_t* dB = &Bs[buf][sOfs[i]];
            dB[0] = pack_bf16(wr[i].x, wr[i].y);
            dB[1] = pack_bf16(wr[i].z, wr[i].w);
        }
    };

    v8f acc[4][2];
    #pragma unroll
    for (int a = 0; a < 4; ++a)
        #pragma unroll
        for (int c = 0; c < 2; ++c)
            acc[a][c] = (v8f)(0.0f);

    auto computeStage = [&](int buf) {
        #pragma unroll
        for (int ks = 0; ks < 2; ++ks) {
            v16bf fa[4], fb[2];
            #pragma unroll
            for (int a2 = 0; a2 < 4; ++a2) {
                int row = wm * 64 + a2 * 16 + lo;
                int cB  = ks * 32 + hi * 16;                 // bf16 column
                const uint32_t* p = &As[buf][row * PITCH];
                u32x4 q0 = *(const u32x4*)(p + (cB >> 1));          // K = hi*16 .. +8
                u32x4 q1 = *(const u32x4*)(p + ((cB + 8) >> 1));    // K = hi*16+8 .. +8
                u32x8 v  = __builtin_shufflevector(q0, q1, 0, 1, 2, 3, 4, 5, 6, 7);
                fa[a2]   = __builtin_bit_cast(v16bf, v);
            }
            #pragma unroll
            for (int c2 = 0; c2 < 2; ++c2) {
                int row = wn * 32 + c2 * 16 + lo;
                int cB  = ks * 32 + hi * 16;                 // K = 16*hi .. +16 contiguous
                const uint32_t* p = &Bs[buf][row * PITCH];
                u32x8 v  = *(const u32x8*)(p + (cB >> 1));
                fb[c2]   = __builtin_bit_cast(v16bf, v);
            }
            #pragma unroll
            for (int a2 = 0; a2 < 4; ++a2)
                #pragma unroll
                for (int c2 = 0; c2 < 2; ++c2)
                    acc[a2][c2] = __builtin_amdgcn_wmma_f32_16x16x32_bf16(
                        false, fa[a2], false, fb[c2],
                        (short)0, acc[a2][c2], false, false);
        }
    };

    // ---- prologue: fill buffer 0
    loadStage(0);
    storeStage(0);
    __syncthreads();

    // ---- fully unrolled double-buffered main loop: one barrier per K-stage,
    //      all global/LDS addresses become base + immediate offset
    #pragma unroll
    for (int kt = 0; kt < 8; ++kt) {
        const int cur = kt & 1;
        if (kt < 7) loadStage((kt + 1) * BK);   // global loads in flight ...
        computeStage(cur);                       // ... while WMMAs run
        if (kt < 7) storeStage(cur ^ 1);
        __syncthreads();
    }

    // ---- epilogue: y = acc*SCALE + b; scatter rows through idx
    #pragma unroll
    for (int c2 = 0; c2 < 2; ++c2) {
        int N = n0 + wn * 32 + c2 * 16 + lo;
        float bias = b[o * 512 + N];                 // LR_MUL == 1
        #pragma unroll
        for (int a2 = 0; a2 < 4; ++a2) {
            int rowg = m0 + wm * 64 + a2 * 16 + hi * 8;   // C/D: M = 8*hi + r
            int kk = rowg / 18;
            int jj = rowg - kk * 18;
            #pragma unroll
            for (int r = 0; r < 8; ++r) {
                int jg = sIdx[jj];
                out[(size_t)kk * KSTR + (size_t)o * OSTR + (size_t)jg * 512 + N]
                    = acc[a2][c2][r] * SCALE + bias;
                if (++jj == 18) { jj = 0; ++kk; }
            }
        }
    }
}

extern "C" void kernel_launch(void* const* d_in, const int* in_sizes, int n_in,
                              void* d_out, int out_size, void* d_ws, size_t ws_size,
                              hipStream_t stream) {
    const float* x  = (const float*)d_in[0];
    const float* W  = (const float*)d_in[1];
    const float* b  = (const float*)d_in[2];
    const int* idx  = (const int*)d_in[3];
    float* out      = (float*)d_out;

    dim3 grid(18 * 36 * 4);   // o * Mtiles * Ntiles = 2592 blocks
    dim3 block(256);          // 8 waves (wave32)
    dirnet_bf16_wmma<<<grid, block, 0, stream>>>(x, W, b, idx, out);
}
